// CINLayer_6588479832455
// MI455X (gfx1250) — matrix operand order repaired and checked by reading, not compiled, not run on hardware
//
#include <hip/hip_runtime.h>

// CIN layer: out[b,c,d] = sum_{h,m} W[c,h*M+m] * xk[b,h,d] * x0[b,m,d] + bias[c]
// = per-b GEMM: out[b] (C x D) = W (C x K) * P[b] (K x D),  K = H*M = 5120,
//   P[b][h*M+m][d] = xk[b,h,d] * x0[b,m,d]  (built on the fly in LDS, bf16)
// Stage 64 K-values per barrier (2 WMMA k-steps), double-buffered.

#define B_SZ 512
#define M_SZ 40
#define H_SZ 128
#define D_SZ 64
#define C_SZ 128
#define K_SZ (H_SZ * M_SZ)     /* 5120 */
#define KSTAGE 64              /* K per staged chunk (2 WMMA k-steps) */
#define NSTAGE (K_SZ / KSTAGE) /* 80 */
#define PSTR 72                /* halves per d-column (padded from 64; 144B, 16B-aligned,
                                  conflict-free b128 column reads) */

typedef __attribute__((ext_vector_type(16))) __bf16 v16bf;
typedef __attribute__((ext_vector_type(8)))  float  v8f;

__device__ __forceinline__ unsigned short f32_to_bf16_rne(float f) {
    union { float f; unsigned u; } v; v.f = f;
    unsigned u = v.u;
    return (unsigned short)((u + 0x7fffu + ((u >> 16) & 1u)) >> 16);
}

__global__ void w_to_bf16_kernel(const float* __restrict__ W,
                                 unsigned short* __restrict__ Wb, int n) {
    int i = blockIdx.x * blockDim.x + threadIdx.x;
    if (i < n) {
        union { float f; unsigned u; } v; v.f = W[i];
        unsigned u = v.u;
        unsigned short r;
        if ((u & 0x7fffffffu) > 0x7f800000u) r = (unsigned short)((u >> 16) | 0x40); // NaN
        else r = (unsigned short)((u + 0x7fffu + ((u >> 16) & 1u)) >> 16);
        Wb[i] = r;
    }
}

__device__ __forceinline__ v16bf load_bfrag(const unsigned short* p) {
    union { uint4 q[2]; v16bf v; } b;
    b.q[0] = *(const uint4*)(p);
    b.q[1] = *(const uint4*)(p + 16);
    return b.v;
}

__launch_bounds__(256, 1)
__global__ void cin_wmma_kernel(const float* __restrict__ x0,
                                const float* __restrict__ xk,
                                const unsigned short* __restrict__ Wb,
                                const float* __restrict__ bias,
                                float* __restrict__ out) {
    __shared__ float xk_s[H_SZ * D_SZ];               // 32 KB
    __shared__ float x0_s[M_SZ * D_SZ];               // 10 KB
    __shared__ unsigned short P[2][D_SZ * PSTR];      // 2 x 9 KB, column-major [d][k]
    __shared__ float bias_s[C_SZ];

    const int b    = blockIdx.x;
    const int tid  = threadIdx.x;
    const int lane = tid & 31;
    const int wave = tid >> 5;       // 0..7, wave owns C rows [wave*16, wave*16+16)
    const int c0   = wave * 16;

    // ---- stage xk[b], x0[b], bias into LDS ----
    const float* xkg = xk + (size_t)b * (H_SZ * D_SZ);
    const float* x0g = x0 + (size_t)b * (M_SZ * D_SZ);
    for (int i = tid; i < H_SZ * D_SZ; i += 256) xk_s[i] = xkg[i];
    for (int i = tid; i < M_SZ * D_SZ; i += 256) x0_s[i] = x0g[i];
    if (tid < C_SZ) bias_s[tid] = bias[tid];

    v8f acc0 = {}, acc1 = {}, acc2 = {}, acc3 = {};

    // P-builder role: thread handles column d_col, rows rgrp*16 .. rgrp*16+15
    const int d_col = tid & 63;
    const int rgrp  = tid >> 6;      // 0..3

    // A-fragment addressing (16-bit A 16x32 ISA layout):
    // lane L: row = L%16; halves 0..7 = K (L<16?0:8)+0..7, halves 8..15 = +16
    const int arow  = c0 + (lane & 15);
    const int akoff = (lane >> 4) << 3;                  // 0 or 8
    const unsigned short* aptr = Wb + (size_t)arow * K_SZ + akoff;

    // B-fragment addressing (16-bit B 32x16 ISA layout):
    // lane L: col = L%16; halves 0..7 = K (L<16?0:8)+0..7, halves 8..15 = +16
    const int bcol = lane & 15;
    const int bk   = (lane >> 4) << 3;                   // 0 or 8

    __syncthreads();

    for (int s = 0; s < NSTAGE; ++s) {
        const int buf   = s & 1;
        const int kbase = s * KSTAGE;

        // ---- build P[buf]: P[d][kk] = xk[h][d] * x0[m][d], k = kbase+kk ----
        {
            int k0 = kbase + (rgrp << 4);
            int h  = k0 / M_SZ;
            int m  = k0 - h * M_SZ;
            union { unsigned short u[8]; uint4 q; } t0, t1;
            #pragma unroll
            for (int j = 0; j < 8; ++j) {
                t0.u[j] = f32_to_bf16_rne(xk_s[h * D_SZ + d_col] * x0_s[m * D_SZ + d_col]);
                if (++m == M_SZ) { m = 0; ++h; }
            }
            #pragma unroll
            for (int j = 0; j < 8; ++j) {
                t1.u[j] = f32_to_bf16_rne(xk_s[h * D_SZ + d_col] * x0_s[m * D_SZ + d_col]);
                if (++m == M_SZ) { m = 0; ++h; }
            }
            unsigned short* pw = &P[buf][d_col * PSTR + (rgrp << 4)];
            *(uint4*)(pw)     = t0.q;
            *(uint4*)(pw + 8) = t1.q;
        }

        // pull next stage's W rows toward this WGP while WMMAs run
        if (s + 1 < NSTAGE) __builtin_prefetch(aptr + kbase + KSTAGE, 0, 1);

        __syncthreads();   // double-buffered: one barrier per stage is sufficient

        // ---- A fragments: W rows for this wave (bf16, L2-resident) ----
        union { uint4 q[2]; v16bf v; } afr0, afr1;
        afr0.q[0] = *(const uint4*)(aptr + kbase);        // K kbase+0..31 slice
        afr0.q[1] = *(const uint4*)(aptr + kbase + 16);
        afr1.q[0] = *(const uint4*)(aptr + kbase + 32);   // K kbase+32..63 slice
        afr1.q[1] = *(const uint4*)(aptr + kbase + 48);

        // ---- B fragments from LDS + 8 WMMAs (4 D-tiles x 2 k-steps) ----
        const unsigned short* pb0 = &P[buf][bcol * PSTR + bk];
        acc0 = __builtin_amdgcn_wmma_f32_16x16x32_bf16(false, afr0.v, false,
                 load_bfrag(pb0 + 0 * 16 * PSTR), (short)0, acc0, false, false);
        acc1 = __builtin_amdgcn_wmma_f32_16x16x32_bf16(false, afr0.v, false,
                 load_bfrag(pb0 + 1 * 16 * PSTR), (short)0, acc1, false, false);
        acc2 = __builtin_amdgcn_wmma_f32_16x16x32_bf16(false, afr0.v, false,
                 load_bfrag(pb0 + 2 * 16 * PSTR), (short)0, acc2, false, false);
        acc3 = __builtin_amdgcn_wmma_f32_16x16x32_bf16(false, afr0.v, false,
                 load_bfrag(pb0 + 3 * 16 * PSTR), (short)0, acc3, false, false);
        acc0 = __builtin_amdgcn_wmma_f32_16x16x32_bf16(false, afr1.v, false,
                 load_bfrag(pb0 + 0 * 16 * PSTR + 32), (short)0, acc0, false, false);
        acc1 = __builtin_amdgcn_wmma_f32_16x16x32_bf16(false, afr1.v, false,
                 load_bfrag(pb0 + 1 * 16 * PSTR + 32), (short)0, acc1, false, false);
        acc2 = __builtin_amdgcn_wmma_f32_16x16x32_bf16(false, afr1.v, false,
                 load_bfrag(pb0 + 2 * 16 * PSTR + 32), (short)0, acc2, false, false);
        acc3 = __builtin_amdgcn_wmma_f32_16x16x32_bf16(false, afr1.v, false,
                 load_bfrag(pb0 + 3 * 16 * PSTR + 32), (short)0, acc3, false, false);
    }

    // ---- store C/D tile (32-bit D 16x16 layout: vgpr i -> row i+8*(lane>=16), col lane%16) ----
    float* outg = out + (size_t)b * (C_SZ * D_SZ);
    const int srow = (lane >> 4) << 3;
    const int scol = lane & 15;
    #pragma unroll
    for (int t = 0; t < 4; ++t) {
        v8f a = (t == 0) ? acc0 : (t == 1) ? acc1 : (t == 2) ? acc2 : acc3;
        #pragma unroll
        for (int i = 0; i < 8; ++i) {
            int c = c0 + srow + i;
            int d = t * 16 + scol;
            outg[c * D_SZ + d] = a[i] + bias_s[c];
        }
    }
}

extern "C" void kernel_launch(void* const* d_in, const int* in_sizes, int n_in,
                              void* d_out, int out_size, void* d_ws, size_t ws_size,
                              hipStream_t stream) {
    (void)in_sizes; (void)n_in; (void)out_size; (void)ws_size;
    const float* x0   = (const float*)d_in[0];   // [B, M, D]
    const float* xk   = (const float*)d_in[1];   // [B, H, D]
    const float* W    = (const float*)d_in[2];   // [C, H*M]
    const float* bias = (const float*)d_in[3];   // [C]
    float* out = (float*)d_out;                  // [B, C, D]

    unsigned short* Wb = (unsigned short*)d_ws;  // bf16 W, 1.31 MB
    const int n = C_SZ * K_SZ;

    hipLaunchKernelGGL(w_to_bf16_kernel, dim3((n + 255) / 256), dim3(256), 0, stream,
                       W, Wb, n);
    hipLaunchKernelGGL(cin_wmma_kernel, dim3(B_SZ), dim3(256), 0, stream,
                       x0, xk, Wb, bias, out);
}
